// MGF_68977174773973
// MI455X (gfx1250) — matrix operand order repaired and checked by reading, not compiled, not run on hardware
//
#include <hip/hip_runtime.h>
#include <hip/hip_bf16.h>

typedef __attribute__((ext_vector_type(16))) __bf16 v16bf;
typedef __attribute__((ext_vector_type(8)))  float  v8f;

// ---------------------------------------------------------------------------
// Prep kernels
// ---------------------------------------------------------------------------

// Repack conv weights (Cout, Cin, taps) fp32 -> WMMA A-fragment order bf16.
// Apack layout: [tap][mTile][kChunk][lane(32)][e(16)], zero padded in M and K.
__global__ void repack_w_kernel(const float* __restrict__ W, __bf16* __restrict__ out,
                                int Cout, int Cin, int taps, int mTiles, int nK) {
  int idx = blockIdx.x * blockDim.x + threadIdx.x;
  int total = taps * mTiles * nK * 512;
  if (idx >= total) return;
  int e    = idx & 15;
  int lane = (idx >> 4) & 31;
  int rest = idx >> 9;
  int kc   = rest % nK;  rest /= nK;
  int mT   = rest % mTiles;
  int tap  = rest / mTiles;
  int M = mT * 16 + (lane & 15);
  int K = kc * 32 + ((lane < 16) ? 0 : 16) + e;
  float v = 0.0f;
  if (M < Cout && K < Cin) v = W[((size_t)M * Cin + K) * taps + tap];
  out[idx] = (__bf16)v;
}

__global__ void pad_bias_kernel(const float* __restrict__ b, float* __restrict__ out,
                                int Cout, int Mp) {
  int i = blockIdx.x * blockDim.x + threadIdx.x;
  if (i < Mp) out[i] = (i < Cout) ? b[i] : 0.0f;
}

// NCHW fp32 -> channels-last bf16 (pad channels zeroed).
__global__ void nchw_to_cl_kernel(const float* __restrict__ X, __bf16* __restrict__ Y,
                                  int C, int Cs, int HW, size_t total) {
  size_t idx = (size_t)blockIdx.x * blockDim.x + threadIdx.x;
  if (idx >= total) return;
  int c = (int)(idx % Cs);
  size_t p = idx / Cs;
  size_t b = p / HW, hw = p % HW;
  float v = (c < C) ? X[(b * C + c) * HW + hw] : 0.0f;
  Y[idx] = (__bf16)v;
}

// ---------------------------------------------------------------------------
// WMMA conv: one wave owns a 16-pixel tile and ALL MT output-channel tiles.
// Each activation B fragment is loaded once and reused for MT WMMAs.
// taps = 1 (1x1 conv) or 9 (3x3 SAME conv via shifted-tap GEMM accumulation).
// ---------------------------------------------------------------------------
template <int MT>
__global__ void wmma_conv_kernel(const __bf16* __restrict__ X,
                                 const __bf16* __restrict__ Apack,
                                 const float* __restrict__ biasPad,
                                 __bf16* __restrict__ Y,
                                 float* __restrict__ YF,      // optional fp32 NCHW out
                                 int CsIn, int nK, int CsOut, int CoutF,
                                 int taps, int relu,
                                 int B, int H, int W) {
  int lane = threadIdx.x & 31;
  int wid  = threadIdx.x >> 5;
  int pixTile = blockIdx.x * (blockDim.x >> 5) + wid;   // wave-uniform
  int p0 = pixTile << 4;              // 16 pixels along x within one row (W%16==0)
  int x0 = p0 % W;
  int y0 = (p0 / W) % H;
  int bb = p0 / (W * H);
  int col = lane & 15;
  int hiK = (lane < 16) ? 0 : 16;     // B fragment: lanes 16..31 hold K=16..31
  v8f acc[MT];
#pragma unroll
  for (int m = 0; m < MT; ++m) acc[m] = (v8f){};

  for (int t = 0; t < taps; ++t) {
    int dy = (taps == 9) ? (t / 3 - 1) : 0;
    int dx = (taps == 9) ? (t % 3 - 1) : 0;
    int ys = y0 + dy;
    if (ys < 0 || ys >= H) continue;  // uniform across wave
    int xs = x0 + col + dx;
    bool ok = (xs >= 0) && (xs < W);
    const __bf16* xrow  = X + ((size_t)(bb * H + ys) * W) * CsIn;
    const __bf16* abase = Apack + ((size_t)t * MT) * nK * 512 + lane * 16;
    for (int kc = 0; kc < nK; ++kc) {
      v16bf bm = {};
      if (ok) bm = *(const v16bf*)(xrow + (size_t)xs * CsIn + kc * 32 + hiK);
#pragma unroll
      for (int m = 0; m < MT; ++m) {
        v16bf a = *(const v16bf*)(abase + ((size_t)m * nK + kc) * 512);
        acc[m] = __builtin_amdgcn_wmma_f32_16x16x32_bf16(false, a, false, bm,
                                                         (short)0, acc[m], false, false);
      }
    }
  }

#pragma unroll
  for (int m = 0; m < MT; ++m) {
    int chBase = m * 16 + ((lane < 16) ? 0 : 8);
#pragma unroll
    for (int v = 0; v < 8; ++v) acc[m][v] += biasPad[chBase + v];
    if (relu) {
#pragma unroll
      for (int v = 0; v < 8; ++v) acc[m][v] = fmaxf(acc[m][v], 0.0f);
    }
    if (YF) {
      size_t hw   = (size_t)y0 * W + x0 + col;
      size_t base = ((size_t)bb * CoutF + chBase) * ((size_t)H * W) + hw;
#pragma unroll
      for (int v = 0; v < 8; ++v) YF[base + (size_t)v * H * W] = acc[m][v];
    } else {
      __bf16* yp = Y + ((size_t)p0 + col) * CsOut + chBase;
#pragma unroll
      for (int v = 0; v < 8; ++v) yp[v] = (__bf16)acc[m][v];
    }
  }
}

// ---------------------------------------------------------------------------
// Elementwise / reduction kernels
// ---------------------------------------------------------------------------
__global__ void ln_kernel(const __bf16* __restrict__ X, __bf16* __restrict__ Y,
                          const float* __restrict__ w, const float* __restrict__ b,
                          int Cs, int C, int P) {
  int p = blockIdx.x * blockDim.x + threadIdx.x;
  if (p >= P) return;
  const __bf16* xp = X + (size_t)p * Cs;
  float mu = 0.0f;
  for (int c = 0; c < C; ++c) mu += (float)xp[c];
  mu /= C;
  float var = 0.0f;
  for (int c = 0; c < C; ++c) { float d = (float)xp[c] - mu; var += d * d; }
  var /= C;
  float inv = rsqrtf(var + 1e-6f);
  __bf16* yp = Y + (size_t)p * Cs;
  for (int c = 0; c < C; ++c)
    yp[c] = (__bf16)(((float)xp[c] - mu) * inv * w[c] + b[c]);
}

__global__ void gate_kernel(const __bf16* __restrict__ X, __bf16* __restrict__ Y,
                            int CsIn, int CsOut, int Ch, size_t total) {
  size_t i = (size_t)blockIdx.x * blockDim.x + threadIdx.x;
  if (i >= total) return;
  int c = (int)(i % Ch);
  size_t p = i / Ch;
  float a = (float)X[p * CsIn + c];
  float b = (float)X[p * CsIn + c + Ch];
  Y[p * CsOut + c] = (__bf16)(a * b);
}

__global__ void dw3x3_kernel(const __bf16* __restrict__ X, const float* __restrict__ Wt,
                             const float* __restrict__ bias, __bf16* __restrict__ Y,
                             int Cs, int C, int B, int H, int W) {
  size_t i = (size_t)blockIdx.x * blockDim.x + threadIdx.x;
  size_t total = (size_t)B * H * W * C;
  if (i >= total) return;
  int c = (int)(i % C);
  size_t p = i / C;
  int x = (int)(p % W), y = (int)((p / W) % H), b = (int)(p / ((size_t)W * H));
  float acc = bias[c];
  for (int t = 0; t < 9; ++t) {
    int ys = y + t / 3 - 1, xs = x + t % 3 - 1;
    if (ys < 0 || ys >= H || xs < 0 || xs >= W) continue;
    size_t ps = ((size_t)(b * H + ys)) * W + xs;
    acc += Wt[c * 9 + t] * (float)X[ps * Cs + c];
  }
  Y[p * Cs + c] = (__bf16)acc;
}

__global__ void pool_kernel(const __bf16* __restrict__ X, float* __restrict__ pool,
                            int Cs, int C, int HW) {
  int bc = blockIdx.x;
  int b = bc / C, c = bc % C;
  const __bf16* base = X + (size_t)b * HW * Cs + c;
  float s = 0.0f;
  for (int i = threadIdx.x; i < HW; i += blockDim.x) s += (float)base[(size_t)i * Cs];
  __shared__ float sm[256];
  sm[threadIdx.x] = s;
  __syncthreads();
  for (int off = 128; off > 0; off >>= 1) {
    if ((int)threadIdx.x < off) sm[threadIdx.x] += sm[threadIdx.x + off];
    __syncthreads();
  }
  if (threadIdx.x == 0) pool[bc] = sm[0] / HW;
}

__global__ void sca_kernel(const float* __restrict__ pool, const float* __restrict__ W,
                           const float* __restrict__ b, float* __restrict__ sca,
                           int C, int B) {
  int i = threadIdx.x;
  if (i >= B * C) return;
  int bb = i / C, c = i % C;
  float acc = b[c];
  for (int k = 0; k < C; ++k) acc += W[c * C + k] * pool[bb * C + k];
  sca[i] = acc;
}

__global__ void scale_bc_kernel(__bf16* __restrict__ X, const float* __restrict__ sca,
                                int Cs, int C, int HW, size_t total) {
  size_t i = (size_t)blockIdx.x * blockDim.x + threadIdx.x;
  if (i >= total) return;
  int c = (int)(i % C);
  size_t p = i / C;
  int b = (int)(p / HW);
  size_t idx = p * Cs + c;
  X[idx] = (__bf16)((float)X[idx] * sca[b * C + c]);
}

// Y = A + B * scale[c]   (residual with per-channel beta/gamma)
__global__ void axpb_kernel(const __bf16* __restrict__ A, const __bf16* __restrict__ Bm,
                            const float* __restrict__ scale, __bf16* __restrict__ Y,
                            int Cs, int C, size_t total) {
  size_t i = (size_t)blockIdx.x * blockDim.x + threadIdx.x;
  if (i >= total) return;
  int c = (int)(i % C);
  size_t p = i / C;
  size_t idx = p * Cs + c;
  Y[idx] = (__bf16)((float)A[idx] + (float)Bm[idx] * scale[c]);
}

// gk*S + gk = gk * (S + 1)
__global__ void gks_kernel(const __bf16* __restrict__ GK, const __bf16* __restrict__ S,
                           __bf16* __restrict__ Y, size_t total) {
  size_t i = (size_t)blockIdx.x * blockDim.x + threadIdx.x;
  if (i >= total) return;
  Y[i] = (__bf16)((float)GK[i] * ((float)S[i] + 1.0f));
}

// fk = tanh(k)/(aff+1e-8); s = sum|fk|+1e-4; k = fk / max(s, 1)
__global__ void getk_kernel(__bf16* __restrict__ K, const float* __restrict__ aff, int P) {
  int p = blockIdx.x * blockDim.x + threadIdx.x;
  if (p >= P) return;
  __bf16* kp = K + (size_t)p * 16;
  float a = aff[0] + 1e-8f;
  float fk[9];
  float s = 1e-4f;
  for (int t = 0; t < 9; ++t) { fk[t] = tanhf((float)kp[t]) / a; s += fabsf(fk[t]); }
  float inv = 1.0f / fmaxf(s, 1.0f);
  for (int t = 0; t < 9; ++t) kp[t] = (__bf16)(fk[t] * inv);
}

// out[p][c] = inp[p][c] + sum_t gk[p][t] * inp[shift_t(p)][c]   (unfold3 einsum)
__global__ void filt_kernel(const __bf16* __restrict__ GK, const __bf16* __restrict__ INP,
                            __bf16* __restrict__ Y, int B, int H, int W) {
  size_t i = (size_t)blockIdx.x * blockDim.x + threadIdx.x;
  size_t total = (size_t)B * H * W * 48;
  if (i >= total) return;
  int c = (int)(i % 48);
  size_t p = i / 48;
  int x = (int)(p % W), y = (int)((p / W) % H), b = (int)(p / ((size_t)W * H));
  float acc = (float)INP[p * 48 + c];
  const __bf16* gp = GK + p * 16;
  for (int t = 0; t < 9; ++t) {
    int ys = y + t / 3 - 1, xs = x + t % 3 - 1;
    if (ys < 0 || ys >= H || xs < 0 || xs >= W) continue;
    size_t ps = ((size_t)(b * H + ys)) * W + xs;
    acc += (float)gp[t] * (float)INP[ps * 48 + c];
  }
  Y[p * 48 + c] = (__bf16)acc;
}

// ---------------------------------------------------------------------------
// Host orchestration
// ---------------------------------------------------------------------------
extern "C" void kernel_launch(void* const* d_in, const int* in_sizes, int n_in,
                              void* d_out, int out_size, void* d_ws, size_t ws_size,
                              hipStream_t stream) {
  (void)in_sizes; (void)n_in; (void)out_size; (void)ws_size;
  const int B = 2, H = 256, W = 256;
  const int HW = H * W;
  const int P = B * HW;

  auto f = [&](int i) { return (const float*)d_in[i]; };
  const float* aff = f(3);

  char* ws = (char*)d_ws;
  size_t off = 0;
  auto alloc = [&](size_t bytes) -> char* {
    char* p = ws + off;
    off = (off + bytes + 255) & ~(size_t)255;
    return p;
  };

  const size_t S48 = (size_t)P * 48 * 2 + 256;
  const size_t S96 = (size_t)P * 96 * 2 + 256;
  const size_t S16 = (size_t)P * 16 * 2 + 256;

  __bf16* clD  = (__bf16*)alloc(S48);   // depth cl; later reused as w_depth
  __bf16* clG  = (__bf16*)alloc(S48);   // guide cl; later reused as w_guide
  __bf16* clS  = (__bf16*)alloc(S16);
  __bf16* bufD = (__bf16*)alloc(S48);   // d
  __bf16* bufG = (__bf16*)alloc(S48);   // g
  __bf16* t48a = (__bf16*)alloc(S48);
  __bf16* t48b = (__bf16*)alloc(S48);
  __bf16* ybuf = (__bf16*)alloc(S48);
  __bf16* t96a = (__bf16*)alloc(S96);
  __bf16* t96b = (__bf16*)alloc(S96);
  __bf16* inpD = (__bf16*)alloc(S48);
  __bf16* inpG = (__bf16*)alloc(S48);
  __bf16* k48  = (__bf16*)alloc(S48);
  __bf16* kA   = (__bf16*)alloc(S16);
  __bf16* kB   = (__bf16*)alloc(S16);
  float* poolb = (float*)alloc(96 * 4);
  float* scab  = (float*)alloc(96 * 4);

  struct WP { __bf16* a; float* bias; int mT; int nK; int taps; };
  auto prep = [&](const float* w, const float* b, int Cout, int Cin, int taps) -> WP {
    int mT = (Cout + 15) / 16, nK = (Cin + 31) / 32;
    size_t n = (size_t)taps * mT * nK * 512;
    __bf16* a = (__bf16*)alloc(n * 2);
    float* bp = (float*)alloc((size_t)mT * 16 * 4);
    repack_w_kernel<<<dim3((unsigned)((n + 255) / 256)), 256, 0, stream>>>(
        w, a, Cout, Cin, taps, mT, nK);
    pad_bias_kernel<<<1, mT * 16, 0, stream>>>(b, bp, Cout, mT * 16);
    return WP{a, bp, mT, nK, taps};
  };

  WP d1  = prep(f(4),  f(5),  48, 48, 9);
  WP g1  = prep(f(6),  f(7),  48, 48, 9);
  WP dk1 = prep(f(8),  f(9),  48, 48, 1);
  WP dk2 = prep(f(10), f(11),  9, 48, 1);
  WP gk1 = prep(f(12), f(13), 48, 48, 1);
  WP gk2 = prep(f(14), f(15),  9, 48, 1);
  WP d2  = prep(f(16), f(17),  9,  9, 1);
  WP g2  = prep(f(18), f(19),  9,  9, 1);
  WP d3  = prep(f(20), f(21), 48, 48, 9);
  WP g3  = prep(f(22), f(23), 48, 48, 9);

  WP iA1 = prep(f(24 + 0),  f(24 + 1),  96, 48, 1);
  WP iA3 = prep(f(24 + 4),  f(24 + 5),  48, 48, 1);
  WP iA4 = prep(f(24 + 8),  f(24 + 9),  96, 48, 1);
  WP iA5 = prep(f(24 + 10), f(24 + 11), 48, 48, 1);
  WP gA1 = prep(f(42 + 0),  f(42 + 1),  96, 48, 1);
  WP gA3 = prep(f(42 + 4),  f(42 + 5),  48, 48, 1);
  WP gA4 = prep(f(42 + 8),  f(42 + 9),  96, 48, 1);
  WP gA5 = prep(f(42 + 10), f(42 + 11), 48, 48, 1);

  auto conv = [&](WP wp, const __bf16* X, int CsIn, __bf16* Y, int CsOut,
                  int relu, float* YF, int CoutF) {
    dim3 grid(P / 16 / 8);          // 8 waves per 256-thread block, 1 tile per wave
    switch (wp.mT) {
      case 1:
        wmma_conv_kernel<1><<<grid, 256, 0, stream>>>(X, wp.a, wp.bias, Y, YF,
            CsIn, wp.nK, CsOut, CoutF, wp.taps, relu, B, H, W);
        break;
      case 3:
        wmma_conv_kernel<3><<<grid, 256, 0, stream>>>(X, wp.a, wp.bias, Y, YF,
            CsIn, wp.nK, CsOut, CoutF, wp.taps, relu, B, H, W);
        break;
      default:
        wmma_conv_kernel<6><<<grid, 256, 0, stream>>>(X, wp.a, wp.bias, Y, YF,
            CsIn, wp.nK, CsOut, CoutF, wp.taps, relu, B, H, W);
        break;
    }
  };

  const unsigned EB = 256;
  auto blocks = [&](size_t n) { return dim3((unsigned)((n + EB - 1) / EB)); };

  // Inputs -> channels-last bf16
  nchw_to_cl_kernel<<<blocks((size_t)P * 48), EB, 0, stream>>>(f(0), clD, 48, 48, HW, (size_t)P * 48);
  nchw_to_cl_kernel<<<blocks((size_t)P * 48), EB, 0, stream>>>(f(1), clG, 48, 48, HW, (size_t)P * 48);
  nchw_to_cl_kernel<<<blocks((size_t)P * 16), EB, 0, stream>>>(f(2), clS, 9, 16, HW, (size_t)P * 16);

  conv(d1, clD, 48, bufD, 48, 0, nullptr, 0);   // d
  conv(g1, clG, 48, bufG, 48, 0, nullptr, 0);   // g

  auto naf = [&](const __bf16* X, int pb, WP c1, WP c3, WP c4, WP c5, __bf16* OUT) {
    ln_kernel<<<blocks((size_t)P), EB, 0, stream>>>(X, t48a, f(pb + 12), f(pb + 13), 48, 48, P);
    conv(c1, t48a, 48, t96a, 96, 0, nullptr, 0);
    dw3x3_kernel<<<blocks((size_t)P * 96), EB, 0, stream>>>(t96a, f(pb + 2), f(pb + 3), t96b, 96, 96, B, H, W);
    gate_kernel<<<blocks((size_t)P * 48), EB, 0, stream>>>(t96b, t48a, 96, 48, 48, (size_t)P * 48);
    pool_kernel<<<96, 256, 0, stream>>>(t48a, poolb, 48, 48, HW);
    sca_kernel<<<1, 96, 0, stream>>>(poolb, f(pb + 6), f(pb + 7), scab, 48, B);
    scale_bc_kernel<<<blocks((size_t)P * 48), EB, 0, stream>>>(t48a, scab, 48, 48, HW, (size_t)P * 48);
    conv(c3, t48a, 48, t48b, 48, 0, nullptr, 0);
    axpb_kernel<<<blocks((size_t)P * 48), EB, 0, stream>>>(X, t48b, f(pb + 16), ybuf, 48, 48, (size_t)P * 48);
    ln_kernel<<<blocks((size_t)P), EB, 0, stream>>>(ybuf, t48a, f(pb + 14), f(pb + 15), 48, 48, P);
    conv(c4, t48a, 48, t96a, 96, 0, nullptr, 0);
    gate_kernel<<<blocks((size_t)P * 48), EB, 0, stream>>>(t96a, t48a, 96, 48, 48, (size_t)P * 48);
    conv(c5, t48a, 48, t48b, 48, 0, nullptr, 0);
    axpb_kernel<<<blocks((size_t)P * 48), EB, 0, stream>>>(ybuf, t48b, f(pb + 17), OUT, 48, 48, (size_t)P * 48);
  };

  auto kchain = [&](const __bf16* src, WP k1, WP k2, WP k3) {
    conv(k1, src, 48, k48, 48, 1, nullptr, 0);                 // relu(conv c->c)
    conv(k2, k48, 48, kA, 16, 0, nullptr, 0);                  // c->9
    gks_kernel<<<blocks((size_t)P * 16), EB, 0, stream>>>(kA, clS, kB, (size_t)P * 16);
    conv(k3, kB, 16, kA, 16, 0, nullptr, 0);                   // 9->9
    getk_kernel<<<blocks((size_t)P), EB, 0, stream>>>(kA, aff, P);
  };

  // Depth branch
  naf(bufD, 24, iA1, iA3, iA4, iA5, inpD);
  kchain(bufG, gk1, gk2, g2);
  filt_kernel<<<blocks((size_t)P * 48), EB, 0, stream>>>(kA, inpD, clD, B, H, W); // w_depth

  // Guide branch
  naf(bufG, 42, gA1, gA3, gA4, gA5, inpG);
  kchain(bufD, dk1, dk2, d2);
  filt_kernel<<<blocks((size_t)P * 48), EB, 0, stream>>>(kA, inpG, clG, B, H, W); // w_guide

  float* outD = (float*)d_out;
  float* outG = outD + (size_t)B * 48 * HW;
  conv(d3, clD, 48, nullptr, 48, 0, outD, 48);
  conv(g3, clG, 48, nullptr, 48, 0, outG, 48);
}
